// TransformerBlock_84473416778448
// MI455X (gfx1250) — matrix operand order repaired and checked by reading, not compiled, not run on hardware
//
#include <hip/hip_runtime.h>
#include <hip/hip_bf16.h>
#include <math.h>

typedef __bf16 bf16;
typedef __attribute__((ext_vector_type(16))) __bf16 v16bf;
typedef __attribute__((ext_vector_type(8)))  __bf16 v8bf;
typedef __attribute__((ext_vector_type(8)))  float  v8f;
typedef int v4i_vs __attribute__((vector_size(16)));   // matches builtin param pointee

union FragAB { v16bf f; v8bf h[2]; };

constexpr int DIMC  = 2048;
constexpr int NH    = 16;
constexpr int HD    = 128;
constexpr int BSZC  = 2;
constexpr int SEQC  = 2048;
constexpr int MROWS = BSZC * SEQC;   // 4096
constexpr int HIDC  = 5632;

// ---------------------------------------------------------------------------
// CDNA5 async global->LDS (ASYNCcnt) path, with compile-safe fallback.
// GLOBAL_LOAD_ASYNC_TO_LDS_B128: each lane copies 16 bytes from global memory into LDS.
// Builtin signature (from hipcc diagnostic): (v4i AS1*, v4i AS3*, imm, cpol).
// ---------------------------------------------------------------------------
#if defined(__HIP_DEVICE_COMPILE__) && \
    __has_builtin(__builtin_amdgcn_global_load_async_to_lds_b128)
#define HAVE_ASYNC_LDS 1
#define ASYNC_CP16(ldst, gsrc)                                              \
  __builtin_amdgcn_global_load_async_to_lds_b128(                           \
      (__attribute__((address_space(1))) v4i_vs*)(gsrc),                    \
      (__attribute__((address_space(3))) v4i_vs*)(ldst), 0, 0)
#if __has_builtin(__builtin_amdgcn_s_wait_asynccnt)
#define WAIT_ASYNC(n) __builtin_amdgcn_s_wait_asynccnt(n)
#else
#define WAIT_ASYNC(n) asm volatile("s_wait_asynccnt %0" ::"n"(n) : "memory")
#endif
#else
#define HAVE_ASYNC_LDS 0
#endif

// ---------------------------------------------------------------------------
// elementwise fp32 -> bf16 cast
// ---------------------------------------------------------------------------
__global__ void cast_f32_bf16(const float* __restrict__ in, bf16* __restrict__ out, size_t n) {
  size_t i = (size_t)blockIdx.x * 256 + threadIdx.x;
  if (i < n) out[i] = (bf16)in[i];
}

// ---------------------------------------------------------------------------
// RMSNorm (one block per row of DIMC) -> bf16 output
// ---------------------------------------------------------------------------
__global__ __launch_bounds__(256)
void rmsnorm_bf16(const float* __restrict__ x, const float* __restrict__ g,
                  bf16* __restrict__ out) {
  __shared__ float red[256];
  const int row = blockIdx.x, tid = threadIdx.x;
  const float* xr = x + (size_t)row * DIMC;
  float s = 0.f;
  for (int i = tid; i < DIMC; i += 256) { float v = xr[i]; s += v * v; }
  red[tid] = s;
  __syncthreads();
  for (int st = 128; st > 0; st >>= 1) {
    if (tid < st) red[tid] += red[tid + st];
    __syncthreads();
  }
  const float r = rsqrtf(red[0] / (float)DIMC + 1e-6f);
  bf16* orow = out + (size_t)row * DIMC;
  for (int i = tid; i < DIMC; i += 256) orow[i] = (bf16)(xr[i] * r * g[i]);
}

// ---------------------------------------------------------------------------
// Tiled WMMA GEMM:  C[M,N] (f32, += res) = A[M,K](bf16) @ B[K,N](bf16)
// BM=BN=128, BK=32; 8 waves (4x2), wave tile 32x64 (8 f32 accumulators).
// Double-buffered LDS tiles; staging via async global->LDS DMA when available,
// else register double-buffering. Fragment layouts per CDNA5 ISA 7.12.2.
// ---------------------------------------------------------------------------
__global__ __launch_bounds__(256)
void gemm_bf16_wmma(const bf16* __restrict__ A, const bf16* __restrict__ B,
                    const float* __restrict__ res, float* __restrict__ C,
                    int M, int N, int K) {
  __shared__ bf16 As[2][128][32];
  __shared__ bf16 Bs[2][32][128];
  const int tid = threadIdx.x, lane = tid & 31, wave = tid >> 5;
  const int wm = wave >> 1, wn = wave & 1;
  const int m0 = blockIdx.y * 128, n0 = blockIdx.x * 128;
  const int kb = (lane & 16) ? 8 : 0;
  // staging geometry: A tile 128x32 (rows ar, ar+64), B tile 32x128 (rows br, br+16)
  const int ar = tid >> 2, ac = (tid & 3) * 8;
  const int br = tid >> 4, bc = (tid & 15) * 8;
  const bf16* a0 = A + (size_t)(m0 + ar) * K + ac;
  const bf16* b0 = B + (size_t)br * N + n0 + bc;
  const size_t aR64 = (size_t)64 * K;
  const size_t bR16 = (size_t)16 * N;
  const size_t bStep = (size_t)32 * N;
  (void)M;
  v8f acc[2][4] = {};

  auto compute = [&](int buf) {
    FragAB a[2], bfr[4];
#pragma unroll
    for (int mt = 0; mt < 2; ++mt) {
      int r = wm * 32 + mt * 16 + (lane & 15);
      a[mt].h[0] = *(const v8bf*)&As[buf][r][kb];
      a[mt].h[1] = *(const v8bf*)&As[buf][r][kb + 16];
    }
#pragma unroll
    for (int nt = 0; nt < 4; ++nt) {
      int c = wn * 64 + nt * 16;
      for (int s = 0; s < 8; ++s) {
        bfr[nt].f[s]     = Bs[buf][kb + s][c + (lane & 15)];
        bfr[nt].f[s + 8] = Bs[buf][kb + 16 + s][c + (lane & 15)];
      }
    }
#pragma unroll
    for (int mt = 0; mt < 2; ++mt)
#pragma unroll
      for (int nt = 0; nt < 4; ++nt)
        acc[mt][nt] = __builtin_amdgcn_wmma_f32_16x16x32_bf16(
            false, a[mt].f, false, bfr[nt].f, (short)0, acc[mt][nt], false, false);
  };

#if HAVE_ASYNC_LDS
  auto stage = [&](int buf, size_t ka, size_t kboff) {
    ASYNC_CP16(&As[buf][ar][ac],      a0 + ka);
    ASYNC_CP16(&As[buf][ar + 64][ac], a0 + aR64 + ka);
    ASYNC_CP16(&Bs[buf][br][bc],      b0 + kboff);
    ASYNC_CP16(&Bs[buf][br + 16][bc], b0 + bR16 + kboff);
  };
  stage(0, 0, 0);
  int buf = 0;
  size_t boff = 0;
  for (int k0 = 0; k0 < K; k0 += 32) {
    if (k0 + 32 < K) {
      stage(buf ^ 1, (size_t)(k0 + 32), boff + bStep);  // DMA next tile
      WAIT_ASYNC(4);                                    // older tile complete
    } else {
      WAIT_ASYNC(0);
    }
    __syncthreads();
    compute(buf);
    __syncthreads();  // readers done before this buffer is re-targeted
    buf ^= 1;
    boff += bStep;
  }
#else
  // register double-buffer: loads for tile k+1 issue before compute of tile k
  v8bf ra0 = *(const v8bf*)a0,        ra1 = *(const v8bf*)(a0 + aR64);
  v8bf rb0 = *(const v8bf*)b0,        rb1 = *(const v8bf*)(b0 + bR16);
  int buf = 0;
  size_t boff = 0;
  for (int k0 = 0; k0 < K; k0 += 32) {
    *(v8bf*)&As[buf][ar][ac]      = ra0;
    *(v8bf*)&As[buf][ar + 64][ac] = ra1;
    *(v8bf*)&Bs[buf][br][bc]      = rb0;
    *(v8bf*)&Bs[buf][br + 16][bc] = rb1;
    __syncthreads();
    if (k0 + 32 < K) {
      boff += bStep;
      ra0 = *(const v8bf*)(a0 + k0 + 32);
      ra1 = *(const v8bf*)(a0 + aR64 + k0 + 32);
      rb0 = *(const v8bf*)(b0 + boff);
      rb1 = *(const v8bf*)(b0 + bR16 + boff);
    }
    compute(buf);
    __syncthreads();
    buf ^= 1;
  }
#endif

#pragma unroll
  for (int mt = 0; mt < 2; ++mt)
#pragma unroll
    for (int nt = 0; nt < 4; ++nt)
#pragma unroll
      for (int j = 0; j < 8; ++j) {
        int rr = m0 + wm * 32 + mt * 16 + j + ((lane & 16) ? 8 : 0);
        int cc = n0 + wn * 64 + nt * 16 + (lane & 15);
        size_t o = (size_t)rr * N + cc;
        float v = acc[mt][nt][j];
        C[o] = res ? res[o] + v : v;
      }
}

// ---------------------------------------------------------------------------
// RoPE + repack: q,k,v f32 [b,s,h*128+d] ->
//   Qb [b,h,s,d] bf16, Ktb [b,h,d,s] bf16 (transposed K), Vb [b,h,s,d] bf16
// ---------------------------------------------------------------------------
__global__ __launch_bounds__(128)
void rope_pack(const float* __restrict__ qf, const float* __restrict__ kf,
               const float* __restrict__ vf, const float* __restrict__ cosb,
               const float* __restrict__ sinb, bf16* __restrict__ Qb,
               bf16* __restrict__ Ktb, bf16* __restrict__ Vb) {
  const int h = blockIdx.x, s = blockIdx.y, b = blockIdx.z, d = threadIdx.x;
  const size_t in = ((size_t)b * SEQC + s) * DIMC + h * HD;
  const float c = cosb[(size_t)s * HD + d], sn = sinb[(size_t)s * HD + d];
  const float q = qf[in + d], k = kf[in + d], v = vf[in + d];
  const float qr = (d < 64) ? -qf[in + d + 64] : qf[in + d - 64];
  const float kr = (d < 64) ? -kf[in + d + 64] : kf[in + d - 64];
  const size_t bh = (size_t)b * NH + h;
  const size_t oq = (bh * SEQC + s) * HD + d;
  Qb[oq] = (bf16)(q * c + qr * sn);
  Ktb[(bh * HD + d) * SEQC + s] = (bf16)(k * c + kr * sn);
  Vb[oq] = (bf16)v;
}

// ---------------------------------------------------------------------------
// Flash attention (causal), one WG per (qblk, h, b); 8 waves x 16 q-rows.
// QK^T and PV via v_wmma_f32_16x16x32_bf16; online softmax via small LDS.
// K/V tiles double-buffered in LDS (async global->LDS DMA when available).
// ---------------------------------------------------------------------------
__global__ __launch_bounds__(256)
void flash_attn_wmma(const bf16* __restrict__ Qb, const bf16* __restrict__ Ktb,
                     const bf16* __restrict__ Vb, bf16* __restrict__ Ob) {
  __shared__ bf16  Ks[2][128][32];   // Kt tile: [d][key]
  __shared__ bf16  Vs[2][32][128];   // V tile:  [key][d]
  __shared__ float Sw[8][16][32];    // per-wave score tile
  __shared__ bf16  Pw[8][16][32];    // per-wave probabilities (bf16 A-frag source)
  __shared__ float mrow[8][16], lrow[8][16], arow[8][16];

  const int tid = threadIdx.x, lane = tid & 31, w = tid >> 5;
  const int qblk = blockIdx.x, h = blockIdx.y, b = blockIdx.z;
  const size_t bh = (size_t)b * NH + h;
  const int hi8 = (lane & 16) ? 8 : 0;
  const int kb  = hi8;               // A-frag K base
  const int qrow = qblk * 128 + w * 16;

  // staging geometry
  const int kd = tid >> 1, khf = (tid & 1) * 16;   // Ks: row d, 16-key half
  const int vr = tid >> 3, vsg = (tid & 7) * 16;   // Vs: key row, 16-d segment
  const bf16* kp0 = Ktb + (bh * HD + kd) * SEQC + khf;
  const bf16* vp0 = Vb + (bh * SEQC + vr) * HD + vsg;

  if (lane < 16) { mrow[w][lane] = -1e30f; lrow[w][lane] = 0.f; }

  FragAB qfr[4];
  {
    const bf16* qp = Qb + (bh * SEQC + qrow + (lane & 15)) * HD;
#pragma unroll
    for (int kc = 0; kc < 4; ++kc) {
      qfr[kc].h[0] = *(const v8bf*)(qp + kc * 32 + kb);
      qfr[kc].h[1] = *(const v8bf*)(qp + kc * 32 + kb + 16);
    }
  }
  v8f o[8] = {};
  const int nk = (qblk + 1) * 4;               // causal pruning
  const float scl = 0.088388347648318447f;     // 1/sqrt(128)
  int buf = 0;

#if HAVE_ASYNC_LDS
  auto stageKV = [&](int bu, int key0) {
    ASYNC_CP16(&Ks[bu][kd][khf],     kp0 + key0);
    ASYNC_CP16(&Ks[bu][kd][khf + 8], kp0 + key0 + 8);
    ASYNC_CP16(&Vs[bu][vr][vsg],     vp0 + (size_t)key0 * HD);
    ASYNC_CP16(&Vs[bu][vr][vsg + 8], vp0 + (size_t)key0 * HD + 8);
  };
  stageKV(0, 0);
#else
  v8bf rk0 = *(const v8bf*)kp0,  rk1 = *(const v8bf*)(kp0 + 8);
  v8bf rv0 = *(const v8bf*)vp0,  rv1 = *(const v8bf*)(vp0 + 8);
#endif

  for (int t = 0; t < nk; ++t) {
    const int key0 = t * 32;
#if HAVE_ASYNC_LDS
    if (t + 1 < nk) {
      stageKV(buf ^ 1, key0 + 32);   // DMA next K/V tile while computing this one
      WAIT_ASYNC(4);
    } else {
      WAIT_ASYNC(0);
    }
    __syncthreads();
#else
    *(v8bf*)&Ks[buf][kd][khf]     = rk0;
    *(v8bf*)&Ks[buf][kd][khf + 8] = rk1;
    *(v8bf*)&Vs[buf][vr][vsg]     = rv0;
    *(v8bf*)&Vs[buf][vr][vsg + 8] = rv1;
    __syncthreads();
    if (t + 1 < nk) {
      rk0 = *(const v8bf*)(kp0 + key0 + 32);
      rk1 = *(const v8bf*)(kp0 + key0 + 40);
      rv0 = *(const v8bf*)(vp0 + (size_t)(key0 + 32) * HD);
      rv1 = *(const v8bf*)(vp0 + (size_t)(key0 + 32) * HD + 8);
    }
#endif

    // S(16x32) = Q(16x128) * K^T(128x32), two 16x16 output tiles
    v8f sc[2] = {};
#pragma unroll
    for (int nt = 0; nt < 2; ++nt)
#pragma unroll
      for (int kc = 0; kc < 4; ++kc) {
        FragAB bk;
        for (int s = 0; s < 8; ++s) {
          bk.f[s]     = Ks[buf][kc * 32 + kb + s][nt * 16 + (lane & 15)];
          bk.f[s + 8] = Ks[buf][kc * 32 + kb + 16 + s][nt * 16 + (lane & 15)];
        }
        sc[nt] = __builtin_amdgcn_wmma_f32_16x16x32_bf16(
            false, qfr[kc].f, false, bk.f, (short)0, sc[nt], false, false);
      }

    // scale + causal mask in C-register layout, spill to LDS
#pragma unroll
    for (int nt = 0; nt < 2; ++nt)
#pragma unroll
      for (int j = 0; j < 8; ++j) {
        int kg = key0 + nt * 16 + (lane & 15);
        int qg = qrow + j + hi8;
        float v = sc[nt][j] * scl + (kg > qg ? -1e30f : 0.f);
        Sw[w][j + hi8][nt * 16 + (lane & 15)] = v;
      }
    __syncthreads();

    // online softmax: lane r (<16) owns q-row r of its wave
    if (lane < 16) {
      float mprev = mrow[w][lane], mx = mprev;
#pragma unroll
      for (int c2 = 0; c2 < 32; ++c2) mx = fmaxf(mx, Sw[w][lane][c2]);
      float al = __expf(mprev - mx), sm = 0.f;
#pragma unroll
      for (int c2 = 0; c2 < 32; ++c2) {
        float p = __expf(Sw[w][lane][c2] - mx);
        sm += p;
        Pw[w][lane][c2] = (bf16)p;
      }
      mrow[w][lane] = mx;
      lrow[w][lane] = lrow[w][lane] * al + sm;
      arow[w][lane] = al;
    }
    __syncthreads();

    // rescale O, then O += P(16x32) * V(32x128)
    float al[8];
#pragma unroll
    for (int j = 0; j < 8; ++j) al[j] = arow[w][j + hi8];
#pragma unroll
    for (int dt = 0; dt < 8; ++dt)
#pragma unroll
      for (int j = 0; j < 8; ++j) o[dt][j] *= al[j];

    FragAB pfr;
    pfr.h[0] = *(const v8bf*)&Pw[w][lane & 15][kb];
    pfr.h[1] = *(const v8bf*)&Pw[w][lane & 15][kb + 16];
#pragma unroll
    for (int dt = 0; dt < 8; ++dt) {
      FragAB bv;
      for (int s = 0; s < 8; ++s) {
        bv.f[s]     = Vs[buf][kb + s][dt * 16 + (lane & 15)];
        bv.f[s + 8] = Vs[buf][kb + 16 + s][dt * 16 + (lane & 15)];
      }
      o[dt] = __builtin_amdgcn_wmma_f32_16x16x32_bf16(
          false, pfr.f, false, bv.f, (short)0, o[dt], false, false);
    }
    __syncthreads();  // PV reads done before next tile re-targets this buffer
    buf ^= 1;
  }

  if (lane < 16) arow[w][lane] = 1.0f / lrow[w][lane];
  __syncthreads();
  float li[8];
#pragma unroll
  for (int j = 0; j < 8; ++j) li[j] = arow[w][j + hi8];
#pragma unroll
  for (int dt = 0; dt < 8; ++dt)
#pragma unroll
    for (int j = 0; j < 8; ++j) {
      size_t row = (size_t)b * SEQC + qrow + j + hi8;
      Ob[row * DIMC + h * HD + dt * 16 + (lane & 15)] = (bf16)(o[dt][j] * li[j]);
    }
}

// ---------------------------------------------------------------------------
// SwiGLU gate: t = silu(u) * g  -> bf16
// ---------------------------------------------------------------------------
__global__ void silu_mul(const float* __restrict__ u, const float* __restrict__ g,
                         bf16* __restrict__ t, size_t n) {
  size_t i = (size_t)blockIdx.x * 256 + threadIdx.x;
  if (i < n) {
    float x = u[i];
    t[i] = (bf16)((x / (1.f + __expf(-x))) * g[i]);
  }
}

// ---------------------------------------------------------------------------
extern "C" void kernel_launch(void* const* d_in, const int* in_sizes, int n_in,
                              void* d_out, int out_size, void* d_ws, size_t ws_size,
                              hipStream_t stream) {
  (void)in_sizes; (void)n_in; (void)out_size; (void)ws_size;
  const float* x    = (const float*)d_in[0];
  const float* cosb = (const float*)d_in[1];
  const float* sinb = (const float*)d_in[2];
  // d_in[3] = additive mask: causal mask applied analytically in the kernel
  const float* wq = (const float*)d_in[4];
  const float* wk = (const float*)d_in[5];
  const float* wv = (const float*)d_in[6];
  const float* wo = (const float*)d_in[7];
  const float* w1 = (const float*)d_in[8];
  const float* w2 = (const float*)d_in[9];
  const float* w3 = (const float*)d_in[10];
  const float* g_attn = (const float*)d_in[11];
  const float* g_ffn  = (const float*)d_in[12];
  float* out = (float*)d_out;

  char* ws = (char*)d_ws;
  size_t off = 0;
  auto alloc = [&](size_t bytes) -> void* {
    void* p = ws + off;
    off = (off + bytes + (size_t)255) & ~(size_t)255;
    return p;
  };

  const size_t DD = (size_t)DIMC * DIMC;
  const size_t DH = (size_t)DIMC * HIDC;
  const size_t MD = (size_t)MROWS * DIMC;
  const size_t MH = (size_t)MROWS * HIDC;

  bf16* wqb = (bf16*)alloc(DD * 2);
  bf16* wkb = (bf16*)alloc(DD * 2);
  bf16* wvb = (bf16*)alloc(DD * 2);
  bf16* wob = (bf16*)alloc(DD * 2);
  bf16* w1b = (bf16*)alloc(DH * 2);
  bf16* w3b = (bf16*)alloc(DH * 2);
  bf16* w2b = (bf16*)alloc(DH * 2);
  bf16* normb = (bf16*)alloc(MD * 2);          // xn, later hn

  char* big = (char*)alloc(2 * MH * 4);        // qf/kf/vf early, U/G later
  float* qf = (float*)big;
  float* kf = (float*)(big + MD * 4);
  float* vf = (float*)(big + 2 * MD * 4);
  float* U  = (float*)big;
  float* G  = (float*)(big + MH * 4);

  char* qkv = (char*)alloc(3 * MD * 2);        // Qb/Ktb/Vb, later Tb (MH*2 fits)
  bf16* Qb  = (bf16*)qkv;
  bf16* Ktb = (bf16*)(qkv + MD * 2);
  bf16* Vb  = (bf16*)(qkv + 2 * MD * 2);
  bf16* Tb  = (bf16*)qkv;

  bf16*  attnout = (bf16*)alloc(MD * 2);
  float* hbuf    = (float*)alloc(MD * 4);

  auto castN = [&](const float* src, bf16* dst, size_t n) {
    cast_f32_bf16<<<dim3((unsigned)((n + 255) / 256)), dim3(256), 0, stream>>>(src, dst, n);
  };
  auto gemm = [&](const bf16* A, const bf16* B, const float* res, float* C,
                  int M, int N, int K) {
    gemm_bf16_wmma<<<dim3(N / 128, M / 128), dim3(256), 0, stream>>>(A, B, res, C, M, N, K);
  };

  // 1) weights -> bf16
  castN(wq, wqb, DD); castN(wk, wkb, DD); castN(wv, wvb, DD); castN(wo, wob, DD);
  castN(w1, w1b, DH); castN(w3, w3b, DH); castN(w2, w2b, DH);

  // 2) attention-branch RMSNorm
  rmsnorm_bf16<<<dim3(MROWS), dim3(256), 0, stream>>>(x, g_attn, normb);

  // 3) QKV projections (WMMA GEMMs)
  gemm(normb, wqb, nullptr, qf, MROWS, DIMC, DIMC);
  gemm(normb, wkb, nullptr, kf, MROWS, DIMC, DIMC);
  gemm(normb, wvb, nullptr, vf, MROWS, DIMC, DIMC);

  // 4) RoPE + head-major repack (K transposed for QK^T fragments)
  rope_pack<<<dim3(NH, SEQC, BSZC), dim3(HD), 0, stream>>>(qf, kf, vf, cosb, sinb,
                                                           Qb, Ktb, Vb);

  // 5) causal flash attention
  flash_attn_wmma<<<dim3(SEQC / 128, NH, BSZC), dim3(256), 0, stream>>>(Qb, Ktb, Vb, attnout);

  // 6) output projection + residual:  h = x + attnout @ wo
  gemm(attnout, wob, x, hbuf, MROWS, DIMC, DIMC);

  // 7) FFN-branch RMSNorm
  rmsnorm_bf16<<<dim3(MROWS), dim3(256), 0, stream>>>(hbuf, g_ffn, normb);

  // 8) W1 / W3 projections
  gemm(normb, w1b, nullptr, U, MROWS, HIDC, DIMC);
  gemm(normb, w3b, nullptr, G, MROWS, HIDC, DIMC);

  // 9) SwiGLU gate
  silu_mul<<<dim3((unsigned)((MH + 255) / 256)), dim3(256), 0, stream>>>(U, G, Tb, MH);

  // 10) W2 projection + residual -> final output
  gemm(Tb, w2b, hbuf, out, MROWS, DIMC, HIDC);
}
